// RaggedHSTUAttn_30270929502463
// MI455X (gfx1250) — matrix operand order repaired and checked by reading, not compiled, not run on hardware
//
#include <hip/hip_runtime.h>

typedef __attribute__((ext_vector_type(16))) __bf16 v16bf;
typedef __attribute__((ext_vector_type(8)))  __bf16 v8bf;
typedef __attribute__((ext_vector_type(4)))  __bf16 v4bf;
typedef __attribute__((ext_vector_type(2)))  __bf16 v2bf;
typedef __attribute__((ext_vector_type(8)))  float  v8f;

constexpr int kH = 4;
constexpr int kD = 128;
constexpr int kN = 2048;
constexpr float kAlpha = 0.08838834764831843f;
constexpr int kLd  = kH * 3 * kD;   // 1536 floats between consecutive tokens
constexpr int kLdo = kH * kD;       // 512 floats out stride
constexpr int kBK  = 32;            // keys per pipeline stage
constexpr int kKpad = 136;          // padded K row (bf16 elems): 272B = 17*16B, bank-spread
constexpr int kVpad = 40;           // padded V^T row: 80B = 5*16B, bank-spread

union U16 { v16bf v; v8bf h[2]; v2bf p[8]; };

__device__ __forceinline__ v2bf cvt2(float a, float b) {
#if __has_builtin(__builtin_amdgcn_cvt_pk_bf16_f32)
  return __builtin_bit_cast(v2bf, __builtin_amdgcn_cvt_pk_bf16_f32(a, b));
#else
  v2bf r; r[0] = (__bf16)a; r[1] = (__bf16)b; return r;
#endif
}

// silu(s) via hardware transcendentals: s * rcp(1 + exp(-s))
__device__ __forceinline__ float silu(float s) {
  return s * __builtin_amdgcn_rcpf(1.f + __expf(-s));
}

// B-operand fragment for Q^T from row-major f32 Q rows (global).
// lane&15 = B column = Q row; lanes 0-15 hold K(=d)=0..15, lanes 16-31 K=16..31.
__device__ __forceinline__ v16bf load_q_frag(const float* base, int lane) {
  const float* r = base + (size_t)(lane & 15) * kLd + ((lane >> 4) << 4);
  U16 u;
#pragma unroll
  for (int i = 0; i < 4; ++i) {
    float4 x = *reinterpret_cast<const float4*>(r + i * 4);
    u.p[2 * i]     = cvt2(x.x, x.y);
    u.p[2 * i + 1] = cvt2(x.z, x.w);
  }
  return u.v;
}

// Cooperative stage of one 32-key block: K as bf16 row-major, V as bf16 transposed.
__device__ __forceinline__ void stage_kv(const float* kp, const float* vp, int nb,
                                         __bf16 (*skb)[kKpad], __bf16 (*svtb)[kVpad],
                                         int t) {
  // --- K: 32 rows x 128 d, thread t does row t>>2, d-quarter (t&3)*32 ---
  {
    const int row = t >> 2;
    const int dq  = (t & 3) << 5;
    const float* src = kp + (size_t)(nb + row) * kLd + dq;
#pragma unroll
    for (int i = 0; i < 8; ++i) {
      float4 x = *reinterpret_cast<const float4*>(src + i * 4);
      union { v4bf v; v2bf p[2]; } u;
      u.p[0] = cvt2(x.x, x.y);
      u.p[1] = cvt2(x.z, x.w);
      *reinterpret_cast<v4bf*>(&skb[row][dq + i * 4]) = u.v;
    }
    if (nb + kBK < kN) __builtin_prefetch(src + (size_t)kBK * kLd, 0, 1);
  }
  // --- V^T: thread t handles rows n0,n0+1 (n0=(t&15)*2), d-chunk (t>>4)*16 ---
  {
    const int n0 = (t & 15) << 1;
    const int db = (t >> 4) << 4;
    const float* s0 = vp + (size_t)(nb + n0) * kLd + db;
    const float* s1 = s0 + kLd;
#pragma unroll
    for (int i = 0; i < 4; ++i) {
      float4 a = *reinterpret_cast<const float4*>(s0 + i * 4);
      float4 b = *reinterpret_cast<const float4*>(s1 + i * 4);
      *reinterpret_cast<v2bf*>(&svtb[db + i * 4 + 0][n0]) = cvt2(a.x, b.x);
      *reinterpret_cast<v2bf*>(&svtb[db + i * 4 + 1][n0]) = cvt2(a.y, b.y);
      *reinterpret_cast<v2bf*>(&svtb[db + i * 4 + 2][n0]) = cvt2(a.z, b.z);
      *reinterpret_cast<v2bf*>(&svtb[db + i * 4 + 3][n0]) = cvt2(a.w, b.w);
    }
    if (nb + kBK < kN) __builtin_prefetch(s0 + (size_t)kBK * kLd, 0, 1);
  }
}

// A-operand (16x32) from bf16 row-major LDS K tile.
// lane&15 = row; lanes 0-15: K(=d) 0..15; lanes 16-31: 16..31 (contiguous).
__device__ __forceinline__ v16bf lds_a_frag(const __bf16 (*skb)[kKpad],
                                            int r0, int c, int lane) {
  const __bf16* rp = &skb[r0 + (lane & 15)][(c << 5) + ((lane >> 4) << 4)];
  U16 u;
  u.h[0] = *reinterpret_cast<const v8bf*>(rp);
  u.h[1] = *reinterpret_cast<const v8bf*>(rp + 8);
  return u.v;
}

// B-operand (32x16) for V from bf16 transposed LDS tile.
// lane&15 = d column; lanes 0-15 hold n={0..7,16..23}, lanes 16-31 n={8..15,24..31}.
__device__ __forceinline__ v16bf lds_v_frag(const __bf16 (*svtb)[kVpad],
                                            int g, int lane) {
  const __bf16* cp = &svtb[(g << 4) + (lane & 15)][(lane >> 4) << 3];
  U16 u;
  u.h[0] = *reinterpret_cast<const v8bf*>(cp);
  u.h[1] = *reinterpret_cast<const v8bf*>(cp + 16);
  return u.v;
}

__global__ __launch_bounds__(128, 1)
void hstu_attn_kernel(const float* __restrict__ qkv,
                      const int* __restrict__ seq_offsets,
                      float* __restrict__ out) {
  __shared__ __bf16 sK[2][kBK][kKpad];   // double-buffered bf16 K tiles
  __shared__ __bf16 sVT[2][kD][kVpad];   // double-buffered bf16 V^T tiles

  const int lane = threadIdx.x & 31;
  const int wave = threadIdx.x >> 5;
  const int h = blockIdx.y;
  const int z = blockIdx.z;
  const int start = seq_offsets[z];
  const int q0 = blockIdx.x * 64 + wave * 16;   // this wave's 16 query rows
  // Wave-uniform copy for scalar branching (keeps EXEC all-1s around WMMA).
  const int q0u = __builtin_amdgcn_readfirstlane(q0);

  const float* qp = qkv + (size_t)(start + q0) * kLd + h * 3 * kD;    // Q rows
  const float* kp = qkv + (size_t)start * kLd + h * 3 * kD + kD;      // K base
  const float* vp = qkv + (size_t)start * kLd + h * 3 * kD + 2 * kD;  // V base

  // Persistent Q^T B-fragments (4 chunks of the D=128 contraction)
  v16bf qf[4];
#pragma unroll
  for (int c = 0; c < 4; ++c) qf[c] = load_q_frag(qp + c * 32, lane);

  const v8f vzero = {0.f, 0.f, 0.f, 0.f, 0.f, 0.f, 0.f, 0.f};
  v8f o[8];
#pragma unroll
  for (int g = 0; g < 8; ++g) o[g] = vzero;

  const int qm = q0 + (lane & 15);        // lane's query index (C-layout column)
  const int krow = (lane >> 4) << 3;      // key-row offset of this lane half
  const int nblocks = 2 * blockIdx.x + 2; // uniform across workgroup (causal max)

  stage_kv(kp, vp, 0, sK[0], sVT[0], threadIdx.x);

  for (int b = 0; b < nblocks; ++b) {
    __syncthreads();
    if (b + 1 < nblocks)
      stage_kv(kp, vp, (b + 1) * kBK, sK[(b + 1) & 1], sVT[(b + 1) & 1],
               threadIdx.x);

    const int nb = b * kBK;
    if (nb <= q0u + 15) {                 // scalar causal skip
      const __bf16 (*skb)[kKpad]  = sK[b & 1];
      const __bf16 (*svtb)[kVpad] = sVT[b & 1];

      // Load all 8 K A-fragments up front so ds latency overlaps the WMMAs.
      v16bf ka0[4], ka1[4];
#pragma unroll
      for (int c = 0; c < 4; ++c) {
        ka0[c] = lds_a_frag(skb, 0, c, lane);
        ka1[c] = lds_a_frag(skb, 16, c, lane);
      }

      // S^T tiles: st0 = K[nb..nb+15] x Q^T, st1 = K[nb+16..nb+31] x Q^T
      v8f st0 = vzero, st1 = vzero;
#pragma unroll
      for (int c = 0; c < 4; ++c)
        st0 = __builtin_amdgcn_wmma_f32_16x16x32_bf16(false, ka0[c], false, qf[c],
                                                      (short)0, st0, false, false);
#pragma unroll
      for (int c = 0; c < 4; ++c)
        st1 = __builtin_amdgcn_wmma_f32_16x16x32_bf16(false, ka1[c], false, qf[c],
                                                      (short)0, st1, false, false);

      // Prefetch all 8 V B-fragments while the silu math runs.
      v16bf vf[8];
#pragma unroll
      for (int g = 0; g < 8; ++g) vf[g] = lds_v_frag(svtb, g, lane);

      // silu + scale on S^T accum (key = nb+krow+j, query = qm); 1/n is
      // hoisted to the epilogue. Mask only the diagonal-crossing block.
      float p0[8], p1[8];
      if (nb + kBK - 1 <= q0u) {          // scalar: block fully causal-valid
#pragma unroll
        for (int j = 0; j < 8; ++j) {
          p0[j] = silu(st0[j] * kAlpha);
          p1[j] = silu(st1[j] * kAlpha);
        }
      } else {
#pragma unroll
        for (int j = 0; j < 8; ++j) {
          float t0 = silu(st0[j] * kAlpha);
          p0[j] = (nb + krow + j <= qm) ? t0 : 0.f;
          float t1 = silu(st1[j] * kAlpha);
          p1[j] = (nb + 16 + krow + j <= qm) ? t1 : 0.f;
        }
      }
      U16 up;
#pragma unroll
      for (int j = 0; j < 4; ++j) {
        up.p[j]     = cvt2(p0[2 * j], p0[2 * j + 1]);  // A elems 0..7
        up.p[4 + j] = cvt2(p1[2 * j], p1[2 * j + 1]);  // A elems 8..15
      }
      v16bf pa = up.v;

      // O(16x128) += P(16x32) x V(32x128), 8 d-groups of 16 columns
#pragma unroll
      for (int g = 0; g < 8; ++g)
        o[g] = __builtin_amdgcn_wmma_f32_16x16x32_bf16(false, pa, false, vf[g],
                                                       (short)0, o[g], false, false);
    }
  }

  // Epilogue: apply 1/n once, store.
  // D tile layout — lane&15 = d col, row = (lane>=16 ? 8 : 0) + vgpr j
  const float inv_n = 1.0f / (float)kN;
  float* op = out + (size_t)(start + q0) * kLdo + h * kD;
  const int mrow = (lane >> 4) << 3;
#pragma unroll
  for (int g = 0; g < 8; ++g) {
#pragma unroll
    for (int j = 0; j < 8; ++j) {
      op[(size_t)(mrow + j) * kLdo + g * 16 + (lane & 15)] = o[g][j] * inv_n;
    }
  }
}

extern "C" void kernel_launch(void* const* d_in, const int* in_sizes, int n_in,
                              void* d_out, int out_size, void* d_ws, size_t ws_size,
                              hipStream_t stream) {
  const float* qkv = (const float*)d_in[0];
  const int* seq_offsets = (const int*)d_in[1];
  float* out = (float*)d_out;
  const int Z = in_sizes[1] - 1;               // seq_offsets has Z+1 entries
  dim3 grid(kN / 64, kH, Z);
  dim3 block(128);                             // 4 waves, 16 query rows each
  hstu_attn_kernel<<<grid, block, 0, stream>>>(qkv, seq_offsets, out);
}
